// MultiHeadAttention_83906481094754
// MI455X (gfx1250) — compile-verified, matrix operand order checked by reading, not run on hardware
//
#include <hip/hip_runtime.h>
#include <hip/hip_bf16.h>

#define D_MODEL 1024
#define N_HEADS 16
#define D_HEAD  64
#define SEQ_L   2048
#define BATCH   2
#define ROWP    (SEQ_L + 64)   // padded LDS score-row stride (bank rotation)

typedef __attribute__((ext_vector_type(16))) __bf16 v16bf;
typedef __attribute__((ext_vector_type(8)))  __bf16 v8bf;
typedef __attribute__((ext_vector_type(4)))  __bf16 v4bf;
typedef __attribute__((ext_vector_type(8)))  float  v8f;

// fp32 -> bf16 (native hardware convert on gfx1250; fuses into v_fma_mix*_bf16)
static __device__ __forceinline__ __bf16 f2bf(float f) {
  return static_cast<__bf16>(f);
}

// K-dim index for element e of a 16-bit WMMA A(16x32)/B(32x16) fragment.
// ISA layout: lanes 0-15 hold K 0-7 (v0-3) and 16-23 (v4-7);
//             lanes 16-31 hold K 8-15 and 24-31.
static __device__ __forceinline__ int kidx(int e, int lane) {
  return ((e & 8) << 1) + ((lane & 16) >> 1) + (e & 7);
}

// ---------------------------------------------------------------------------
// Elementwise fp32 -> bf16 (pre-convert activations & weights once).
// ---------------------------------------------------------------------------
__launch_bounds__(256)
__global__ void cvt_f32_bf16(const float* __restrict__ src,
                             __bf16* __restrict__ dst, int n) {
  const int i = (blockIdx.x * 256 + threadIdx.x) * 4;
  if (i < n) {
    const float4 f = *(const float4*)(src + i);
    v4bf o = {f2bf(f.x), f2bf(f.y), f2bf(f.z), f2bf(f.w)};
    *(v4bf*)(dst + i) = o;
  }
}

// ---------------------------------------------------------------------------
// Y = A(MxK bf16) @ W^T(NxK bf16) + bias, via v_wmma_f32_16x16x32_bf16.
// MODE 0: store bf16 into head-major [B, H, L, D_HEAD] buffer (QKV proj)
// MODE 1: store fp32 row-major [M, N] (output projection)
// Block: 128 threads (4 waves); each wave: 32x64 tile, 8 independent WMMAs/K.
// Grid: (M/128, N/64).
// ---------------------------------------------------------------------------
template <int MODE>
__launch_bounds__(128)
__global__ void proj_gemm_wmma(const __bf16* __restrict__ A,
                               const __bf16* __restrict__ W,
                               const float* __restrict__ bias,
                               __bf16* __restrict__ outBF,
                               float* __restrict__ outF) {
  const int lane = threadIdx.x & 31;
  const int wave = threadIdx.x >> 5;
  const int m0 = blockIdx.x * 128 + wave * 32;
  const int n0 = blockIdx.y * 64;

  v8f acc0[4] = {};
  v8f acc1[4] = {};

  const __bf16* arow0 = A + (m0 + (lane & 15)) * D_MODEL;
  const __bf16* arow1 = arow0 + 16 * D_MODEL;

  for (int kk = 0; kk < D_MODEL; kk += 32) {
    v16bf a0, a1;
#pragma unroll
    for (int e = 0; e < 16; ++e) {
      a0[e] = arow0[kk + kidx(e, lane)];
      a1[e] = arow1[kk + kidx(e, lane)];
    }
    if (kk + 32 < D_MODEL) {
      __builtin_prefetch(arow0 + kk + 32, 0, 0);  // global_prefetch_b8
      __builtin_prefetch(arow1 + kk + 32, 0, 0);
    }
#pragma unroll
    for (int t = 0; t < 4; ++t) {
      const __bf16* wrow = W + (n0 + t * 16 + (lane & 15)) * D_MODEL + kk;
      v16bf bm;
#pragma unroll
      for (int e = 0; e < 16; ++e) bm[e] = wrow[kidx(e, lane)];
      acc0[t] = __builtin_amdgcn_wmma_f32_16x16x32_bf16(
          false, a0, false, bm, (short)0, acc0[t], false, false);
      acc1[t] = __builtin_amdgcn_wmma_f32_16x16x32_bf16(
          false, a1, false, bm, (short)0, acc1[t], false, false);
    }
  }

#pragma unroll
  for (int t = 0; t < 4; ++t) {
    const int col = n0 + t * 16 + (lane & 15);
    const float bv = bias[col];
#pragma unroll
    for (int g = 0; g < 2; ++g) {
#pragma unroll
      for (int r = 0; r < 8; ++r) {
        const int row = m0 + g * 16 + r + ((lane >> 4) << 3);
        const float v = (g ? acc1[t][r] : acc0[t][r]) + bv;
        if (MODE == 0) {
          const int bb = row >> 11;            // L = 2048
          const int l  = row & (SEQ_L - 1);
          const int h  = col >> 6;             // d_h = 64
          const int d  = col & (D_HEAD - 1);
          outBF[(((long)(bb * N_HEADS + h) * SEQ_L) + l) * D_HEAD + d] = f2bf(v);
        } else {
          outF[(long)row * D_MODEL + col] = v;
        }
      }
    }
  }
}

// ---------------------------------------------------------------------------
// Attention for one 16-row query strip of one (b, h).
// LDS: bf16 score strip [16][ROWP] (padded) + fp32 reduction scratch.
// Phase 1: Q K^T/8 (x2-unrolled independent WMMA chains).
// Phase 2: vectorized softmax (v8bf LDS loads, float4 attn stores).
// Phase 3: ctx = P @ V with dual accumulators.
// Block: 128 threads (4 waves). Grid: (L/16, H, B).
// ---------------------------------------------------------------------------
__launch_bounds__(128)
__global__ void attn_wmma(const __bf16* __restrict__ Qb,
                          const __bf16* __restrict__ Kb,
                          const __bf16* __restrict__ Vb,
                          float* __restrict__ attn_out,
                          __bf16* __restrict__ ctx) {
  extern __shared__ char smem_raw[];
  __bf16* S  = (__bf16*)smem_raw;                        // [16][ROWP]
  float* red = (float*)(smem_raw + 16 * ROWP * 2);       // [16][8]

  const int lane = threadIdx.x & 31;
  const int wave = threadIdx.x >> 5;
  const int q0 = blockIdx.x * 16;
  const int h  = blockIdx.y;
  const int bb = blockIdx.z;
  const long headBase = (long)(bb * N_HEADS + h) * SEQ_L;

  // Q fragment: rows q0..q0+15, d_h=64 -> two K=32 chunks (live all phase 1)
  const __bf16* qptr = Qb + (headBase + q0 + (lane & 15)) * D_HEAD;
  v16bf aq0, aq1;
#pragma unroll
  for (int e = 0; e < 16; ++e) {
    aq0[e] = qptr[kidx(e, lane)];
    aq1[e] = qptr[32 + kidx(e, lane)];
  }

  // ---- phase 1: score strip into LDS (two independent tiles per iter) -----
  for (int j = wave; j < SEQ_L / 16; j += 8) {
    const __bf16* kp0 = Kb + (headBase + j * 16 + (lane & 15)) * D_HEAD;
    const __bf16* kp1 = kp0 + 4 * 16 * D_HEAD;           // tile j+4
    v16bf bk00, bk01, bk10, bk11;
#pragma unroll
    for (int e = 0; e < 16; ++e) {
      const int k0 = kidx(e, lane);
      bk00[e] = kp0[k0];       bk01[e] = kp0[32 + k0];
      bk10[e] = kp1[k0];       bk11[e] = kp1[32 + k0];
    }
    v8f s0 = {}, s1 = {};
    s0 = __builtin_amdgcn_wmma_f32_16x16x32_bf16(false, aq0, false, bk00, (short)0, s0, false, false);
    s1 = __builtin_amdgcn_wmma_f32_16x16x32_bf16(false, aq0, false, bk10, (short)0, s1, false, false);
    s0 = __builtin_amdgcn_wmma_f32_16x16x32_bf16(false, aq1, false, bk01, (short)0, s0, false, false);
    s1 = __builtin_amdgcn_wmma_f32_16x16x32_bf16(false, aq1, false, bk11, (short)0, s1, false, false);
#pragma unroll
    for (int r = 0; r < 8; ++r) {
      const int rr = (r + ((lane >> 4) << 3)) * ROWP + (lane & 15);
      S[rr + j * 16]       = f2bf(s0[r] * 0.125f);
      S[rr + (j + 4) * 16] = f2bf(s1[r] * 0.125f);
    }
  }
  __syncthreads();

  // ---- phase 2: vectorized row softmax ------------------------------------
  // thread t owns 16B chunks idx = sub + 8*i of row (t>>3); 8 subs cover
  // 256 contiguous bytes per step -> coalesced LDS b128 / global b128.
  const int row = threadIdx.x >> 3;
  const int sub = threadIdx.x & 7;
  const v8bf* Sv = (const v8bf*)(S + row * ROWP);

  float mx = -3.402823e38f;
#pragma unroll 4
  for (int i = 0; i < SEQ_L / 64; ++i) {
    const v8bf x = Sv[sub + 8 * i];
#pragma unroll
    for (int e = 0; e < 8; ++e) mx = fmaxf(mx, (float)x[e]);
  }
  red[row * 8 + sub] = mx;
  __syncthreads();
  mx = red[row * 8];
#pragma unroll
  for (int i = 1; i < 8; ++i) mx = fmaxf(mx, red[row * 8 + i]);
  __syncthreads();  // done reading red before reuse

  float ps = 0.0f;
#pragma unroll 4
  for (int i = 0; i < SEQ_L / 64; ++i) {
    const v8bf x = Sv[sub + 8 * i];
#pragma unroll
    for (int e = 0; e < 8; ++e) ps += __expf((float)x[e] - mx);
  }
  red[row * 8 + sub] = ps;
  __syncthreads();
  float tot = 0.0f;
#pragma unroll
  for (int i = 0; i < 8; ++i) tot += red[row * 8 + i];
  const float inv = 1.0f / tot;

  float* aout = attn_out + (headBase + q0 + row) * SEQ_L;
  v8bf* Svw = (v8bf*)(S + row * ROWP);
#pragma unroll 2
  for (int i = 0; i < SEQ_L / 64; ++i) {
    const int idx = sub + 8 * i;
    const v8bf x = Svw[idx];
    float p[8];
#pragma unroll
    for (int e = 0; e < 8; ++e) p[e] = __expf((float)x[e] - mx) * inv;
    float4 lo = {p[0], p[1], p[2], p[3]};
    float4 hi = {p[4], p[5], p[6], p[7]};
    *(float4*)(aout + idx * 8)     = lo;   // fp32 attn, global_store_b128
    *(float4*)(aout + idx * 8 + 4) = hi;
    v8bf pb;
#pragma unroll
    for (int e = 0; e < 8; ++e) pb[e] = f2bf(p[e]);
    Svw[idx] = pb;                         // bf16 P for the PV matmul
  }
  __syncthreads();

  // ---- phase 3: ctx = P @ V  (dual accumulators, wave owns 16 cols) -------
  const int ncol = wave * 16 + (lane & 15);
  v8f cacc0 = {}, cacc1 = {};
  const __bf16* Srow = S + (lane & 15) * ROWP;
  for (int kb = 0; kb < SEQ_L; kb += 64) {
    v16bf pa0, pa1, vb0, vb1;
#pragma unroll
    for (int e = 0; e < 16; ++e) {
      const int k0 = kidx(e, lane);
      pa0[e] = Srow[kb + k0];
      pa1[e] = Srow[kb + 32 + k0];
      vb0[e] = Vb[(headBase + kb + k0) * D_HEAD + ncol];
      vb1[e] = Vb[(headBase + kb + 32 + k0) * D_HEAD + ncol];
    }
    cacc0 = __builtin_amdgcn_wmma_f32_16x16x32_bf16(
        false, pa0, false, vb0, (short)0, cacc0, false, false);
    cacc1 = __builtin_amdgcn_wmma_f32_16x16x32_bf16(
        false, pa1, false, vb1, (short)0, cacc1, false, false);
  }
  const v8f cacc = cacc0 + cacc1;
#pragma unroll
  for (int r = 0; r < 8; ++r) {
    const int row2 = q0 + r + ((lane >> 4) << 3);
    ctx[((long)bb * SEQ_L + row2) * D_MODEL + h * D_HEAD + ncol] = f2bf(cacc[r]);
  }
}

// ---------------------------------------------------------------------------
extern "C" void kernel_launch(void* const* d_in, const int* in_sizes, int n_in,
                              void* d_out, int out_size, void* d_ws, size_t ws_size,
                              hipStream_t stream) {
  (void)in_sizes; (void)n_in; (void)out_size; (void)ws_size;

  const float* q   = (const float*)d_in[0];
  const float* k   = (const float*)d_in[1];
  const float* v   = (const float*)d_in[2];
  const float* w_q = (const float*)d_in[3];
  const float* b_q = (const float*)d_in[4];
  const float* w_k = (const float*)d_in[5];
  const float* b_k = (const float*)d_in[6];
  const float* w_v = (const float*)d_in[7];
  const float* b_v = (const float*)d_in[8];
  const float* w_o = (const float*)d_in[9];
  const float* b_o = (const float*)d_in[10];

  const int NA = BATCH * SEQ_L * D_MODEL;  // 4 Mi elements (activations)
  const int NW = D_MODEL * D_MODEL;        // 1 Mi elements (weights)

  // workspace layout (bf16):
  //  Aq/Ak/Av row-major (3x8MB) | Wq/Wk/Wv/Wo (4x2MB) |
  //  Qb/Kb/Vb head-major (3x8MB) | CTX row-major (8MB)      total 64 MB
  char* ws = (char*)d_ws;
  const size_t ABYT = (size_t)NA * sizeof(__bf16);   // 8 MB
  const size_t WBYT = (size_t)NW * sizeof(__bf16);   // 2 MB
  __bf16* Aq = (__bf16*)(ws);
  __bf16* Ak = (__bf16*)(ws + ABYT);
  __bf16* Av = (__bf16*)(ws + 2 * ABYT);
  __bf16* Wq = (__bf16*)(ws + 3 * ABYT);
  __bf16* Wk = (__bf16*)(ws + 3 * ABYT + WBYT);
  __bf16* Wv = (__bf16*)(ws + 3 * ABYT + 2 * WBYT);
  __bf16* Wo = (__bf16*)(ws + 3 * ABYT + 3 * WBYT);
  __bf16* Qb = (__bf16*)(ws + 3 * ABYT + 4 * WBYT);
  __bf16* Kb = (__bf16*)(ws + 4 * ABYT + 4 * WBYT);
  __bf16* Vb = (__bf16*)(ws + 5 * ABYT + 4 * WBYT);
  __bf16* CTX = (__bf16*)(ws + 6 * ABYT + 4 * WBYT);

  float* out  = (float*)d_out;                          // [B,L,D]
  float* attn = out + (size_t)BATCH * SEQ_L * D_MODEL;  // [B,H,L,L]

  // --- pre-convert activations & weights to bf16 ---
  const dim3 bC(256, 1, 1);
  cvt_f32_bf16<<<dim3(NA / 1024, 1, 1), bC, 0, stream>>>(q, Aq, NA);
  cvt_f32_bf16<<<dim3(NA / 1024, 1, 1), bC, 0, stream>>>(k, Ak, NA);
  cvt_f32_bf16<<<dim3(NA / 1024, 1, 1), bC, 0, stream>>>(v, Av, NA);
  cvt_f32_bf16<<<dim3(NW / 1024, 1, 1), bC, 0, stream>>>(w_q, Wq, NW);
  cvt_f32_bf16<<<dim3(NW / 1024, 1, 1), bC, 0, stream>>>(w_k, Wk, NW);
  cvt_f32_bf16<<<dim3(NW / 1024, 1, 1), bC, 0, stream>>>(w_v, Wv, NW);
  cvt_f32_bf16<<<dim3(NW / 1024, 1, 1), bC, 0, stream>>>(w_o, Wo, NW);

  // --- QKV projections ---
  const dim3 gG(BATCH * SEQ_L / 128, D_MODEL / 64, 1);
  const dim3 bG(128, 1, 1);
  proj_gemm_wmma<0><<<gG, bG, 0, stream>>>(Aq, Wq, b_q, Qb, (float*)nullptr);
  proj_gemm_wmma<0><<<gG, bG, 0, stream>>>(Ak, Wk, b_k, Kb, (float*)nullptr);
  proj_gemm_wmma<0><<<gG, bG, 0, stream>>>(Av, Wv, b_v, Vb, (float*)nullptr);

  // --- attention ---
  const dim3 gA(SEQ_L / 16, N_HEADS, BATCH);
  const size_t shm = 16 * ROWP * sizeof(__bf16) + 16 * 8 * sizeof(float);
  attn_wmma<<<gA, dim3(128, 1, 1), shm, stream>>>(Qb, Kb, Vb, attn, CTX);

  // --- output projection ---
  proj_gemm_wmma<1><<<gG, bG, 0, stream>>>(CTX, Wo, b_o, (__bf16*)nullptr, out);
}
